// GPT3Attention_46823733461028
// MI455X (gfx1250) — compile-verified, matrix operand order checked by reading, not run on hardware
//
#include <hip/hip_runtime.h>

// ---------------------------------------------------------------------------
// GPT-3 style attention block for MI455X (gfx1250), bf16 WMMA everywhere.
// B=2, S=4096, D=768, H=12, depth=64.
// Round 2: 2x2 register tiling in GEMMs, 2 Q-tiles per wave in flash attn
// (doubles FLOP/byte in both compute-bound stages).
// ---------------------------------------------------------------------------

#define B_  2
#define S_  4096
#define D_  768
#define H_  12
#define DEP 64
#define M_  (B_ * S_)            // 8192 rows for the projection GEMMs

typedef __attribute__((ext_vector_type(16))) __bf16 v16bf;
typedef __attribute__((ext_vector_type(8)))  float  v8f;

// D = A(16x32 bf16) * B(32x16 bf16) + C(16x16 f32)
__device__ __forceinline__ v8f wmma_bf16(v16bf a, v16bf b, v8f c) {
    return __builtin_amdgcn_wmma_f32_16x16x32_bf16(
        /*neg_a=*/false, a, /*neg_b=*/false, b,
        /*c_mod=*/(short)0, c, /*reuse_a=*/false, /*reuse_b=*/false);
}

// Load a 16x32 fragment whose K-dimension is contiguous in memory.
// ISA 7.12.2 16-bit layout: lanes 0-15 hold K = {0..7,16..23} pairs,
// lanes 16-31 hold K = {8..15,24..31} pairs, dword-packed.
__device__ __forceinline__ v16bf load_km(const __bf16* base, int ld, int lane, int k0) {
    const int g = lane >> 4;
    const int m = lane & 15;
    const __bf16* p = base + (size_t)m * ld + k0 + 8 * g;
    v16bf f;
#pragma unroll
    for (int v = 0; v < 8; ++v) {
        const int kb = (v < 4) ? (2 * v) : (8 + 2 * v);   // 0,2,4,6,16,18,20,22
        f[2 * v]     = p[kb];
        f[2 * v + 1] = p[kb + 1];
    }
    return f;
}

// ---------------------------------------------------------------------------
// Elementwise f32 -> bf16
// ---------------------------------------------------------------------------
__global__ void k_cvt_bf16(const float* __restrict__ in, __bf16* __restrict__ out, int n) {
    int i = blockIdx.x * blockDim.x + threadIdx.x;
    if (i < n) out[i] = (__bf16)in[i];
}

// W[k][n] f32  ->  Wt[n][k] bf16   (so B fragments load K-contiguously)
__global__ void k_w_transpose_bf16(const float* __restrict__ W, __bf16* __restrict__ Wt) {
    int i = blockIdx.x * blockDim.x + threadIdx.x;
    if (i >= D_ * D_) return;
    int n = i % D_;
    int k = i / D_;
    Wt[(size_t)n * D_ + k] = (__bf16)W[(size_t)k * D_ + n];
}

// ---------------------------------------------------------------------------
// 16x16 C-tile store helper
//   mode 0: dst bf16 head-split [bh][s][dep]              (Q, K)
//   mode 1: dst bf16 head-split transposed [bh][dep][s]   (V)
//   mode 2: dst f32 flat [M_][D_]                          (output proj)
// ---------------------------------------------------------------------------
__device__ __forceinline__ void store_tile(v8f c, int row0, int col0, int lane,
                                           const float* __restrict__ bias,
                                           __bf16* __restrict__ dstb,
                                           float* __restrict__ dstf, int mode) {
    const int g  = lane >> 4;
    const int ci = lane & 15;
    const int col = col0 + ci;
    const float bv = bias[col];
    const int h = col >> 6, dep = col & 63;
#pragma unroll
    for (int r = 0; r < 8; ++r) {
        const int row = row0 + r + 8 * g;
        const float val = c[r] + bv;
        if (mode == 2) {
            dstf[(size_t)row * D_ + col] = val;
        } else {
            const int b_ = row / S_, s = row % S_;
            size_t o;
            if (mode == 0)
                o = ((size_t)(b_ * H_ + h) * S_ + s) * DEP + dep;   // [bh][s][dep]
            else
                o = ((size_t)(b_ * H_ + h) * DEP + dep) * S_ + s;   // [bh][dep][s]
            dstb[o] = (__bf16)val;
        }
    }
}

// ---------------------------------------------------------------------------
// GEMM: out = X(bf16 [M_ x D_]) * W^T(bf16 [D_ x D_], n-major) + b
// 4 waves per block; each wave owns a 32x32 output tile (2x2 register tile):
// 4 WMMAs per k-step reuse 2 A-frags + 2 B-frags -> 2x FLOP/byte vs 16x16.
// ---------------------------------------------------------------------------
__global__ __launch_bounds__(128) void k_gemm(const __bf16* __restrict__ A,
                                              const __bf16* __restrict__ Bt,
                                              const float* __restrict__ bias,
                                              __bf16* __restrict__ dstb,
                                              float* __restrict__ dstf,
                                              int mode) {
    const int lane = threadIdx.x & 31;
    const int wave = threadIdx.x >> 5;
    const int ntn  = D_ / 32;                 // 24 tiles along N
    const int tile = blockIdx.x * 4 + wave;
    const int mt = tile / ntn, nt = tile % ntn;
    const int row0 = mt * 32, col0 = nt * 32;

    v8f c00 = {}, c01 = {}, c10 = {}, c11 = {};
#pragma unroll 2
    for (int k0 = 0; k0 < D_; k0 += 32) {
        const v16bf a0 = load_km(A  + (size_t)row0 * D_,        D_, lane, k0);
        const v16bf a1 = load_km(A  + (size_t)(row0 + 16) * D_, D_, lane, k0);
        const v16bf b0 = load_km(Bt + (size_t)col0 * D_,        D_, lane, k0);
        const v16bf b1 = load_km(Bt + (size_t)(col0 + 16) * D_, D_, lane, k0);
        c00 = wmma_bf16(a0, b0, c00);
        c01 = wmma_bf16(a0, b1, c01);
        c10 = wmma_bf16(a1, b0, c10);
        c11 = wmma_bf16(a1, b1, c11);
    }

    store_tile(c00, row0,      col0,      lane, bias, dstb, dstf, mode);
    store_tile(c01, row0,      col0 + 16, lane, bias, dstb, dstf, mode);
    store_tile(c10, row0 + 16, col0,      lane, bias, dstb, dstf, mode);
    store_tile(c11, row0 + 16, col0 + 16, lane, bias, dstb, dstf, mode);
}

// ---------------------------------------------------------------------------
// Flash attention: 1 wave per (32-query block, bh): two 16-row Q tiles share
// every K/V fragment. Per 32-key step: 8 WMMA for Q*K^T, online softmax with
// shfl_xor row reductions, P relayout via LDS, 8 WMMA for P*V
// (V pre-transposed [bh][dep][S]).
// ---------------------------------------------------------------------------
__global__ __launch_bounds__(32) void k_flash(const __bf16* __restrict__ Qh,
                                              const __bf16* __restrict__ Kh,
                                              const __bf16* __restrict__ Vt,
                                              const float*  __restrict__ mask,
                                              __bf16* __restrict__ Attn) {
    __shared__ __bf16 pLDS[2][16][40];        // two 16x32 P tiles, padded

    const int lane = threadIdx.x & 31;
    const int qb32 = blockIdx.x;              // 32-query block index
    const int bh = blockIdx.y;
    const int b_ = bh / H_, h = bh % H_;
    const int g  = lane >> 4;
    const int ci = lane & 15;

    const __bf16* Kbase = Kh + (size_t)bh * S_ * DEP;
    const __bf16* Vbase = Vt + (size_t)bh * DEP * S_;

    v16bf qa[2][2];
    int q0[2];
#pragma unroll
    for (int t = 0; t < 2; ++t) {
        q0[t] = qb32 * 32 + t * 16;
        const __bf16* Qbase = Qh + ((size_t)bh * S_ + q0[t]) * DEP;
        qa[t][0] = load_km(Qbase, DEP, lane, 0);
        qa[t][1] = load_km(Qbase, DEP, lane, 32);
    }

    v8f o[2][4] = {};
    float m[2][8], l[2][8];
#pragma unroll
    for (int t = 0; t < 2; ++t)
#pragma unroll
        for (int r = 0; r < 8; ++r) { m[t][r] = -1e30f; l[t][r] = 0.0f; }
    const float scale = 0.125f;               // 1/sqrt(64)

    for (int key0 = 0; key0 < S_; key0 += 32) {
        // ---- shared K fragments for this 32-key step ----
        v16bf kf[4];
        kf[0] = load_km(Kbase + (size_t)key0 * DEP,        DEP, lane, 0);
        kf[1] = load_km(Kbase + (size_t)key0 * DEP,        DEP, lane, 32);
        kf[2] = load_km(Kbase + (size_t)(key0 + 16) * DEP, DEP, lane, 0);
        kf[3] = load_km(Kbase + (size_t)(key0 + 16) * DEP, DEP, lane, 32);
        if (key0 + 32 < S_)
            __builtin_prefetch(Kbase + (size_t)(key0 + 32) * DEP, 0, 3);

#pragma unroll
        for (int t = 0; t < 2; ++t) {
            // ---- logits: S[16 x 32] = Q(16x64) * K^T(64x32) ----
            v8f s0 = {}, s1 = {};
            s0 = wmma_bf16(qa[t][0], kf[0], s0);
            s0 = wmma_bf16(qa[t][1], kf[1], s0);
            s1 = wmma_bf16(qa[t][0], kf[2], s1);
            s1 = wmma_bf16(qa[t][1], kf[3], s1);

            // ---- mask + online softmax ----
            float x0[8], x1[8], rm[8];
#pragma unroll
            for (int r = 0; r < 8; ++r) {
                const int row = q0[t] + r + 8 * g;
                const float m0 = mask[(size_t)row * S_ + key0 + ci];
                const float m1 = mask[(size_t)row * S_ + key0 + 16 + ci];
                x0[r] = s0[r] * scale + m0 * (-1e9f);
                x1[r] = s1[r] * scale + m1 * (-1e9f);
                rm[r] = fmaxf(x0[r], x1[r]);
            }
#pragma unroll
            for (int d = 1; d < 16; d <<= 1)
#pragma unroll
                for (int r = 0; r < 8; ++r)
                    rm[r] = fmaxf(rm[r], __shfl_xor(rm[r], d, 32));

            float alpha[8], rs[8];
#pragma unroll
            for (int r = 0; r < 8; ++r) {
                const float mn = fmaxf(m[t][r], rm[r]);
                alpha[r] = __expf(m[t][r] - mn);
                m[t][r] = mn;
                x0[r] = __expf(x0[r] - mn);
                x1[r] = __expf(x1[r] - mn);
                rs[r] = x0[r] + x1[r];
            }
#pragma unroll
            for (int d = 1; d < 16; d <<= 1)
#pragma unroll
                for (int r = 0; r < 8; ++r)
                    rs[r] += __shfl_xor(rs[r], d, 32);

#pragma unroll
            for (int r = 0; r < 8; ++r) {
                l[t][r] = l[t][r] * alpha[r] + rs[r];
                o[t][0][r] *= alpha[r]; o[t][1][r] *= alpha[r];
                o[t][2][r] *= alpha[r]; o[t][3][r] *= alpha[r];
                // stash P (C-fragment layout) into LDS for A-fragment reload
                pLDS[t][r + 8 * g][ci]      = (__bf16)x0[r];
                pLDS[t][r + 8 * g][16 + ci] = (__bf16)x1[r];
            }
        }
        __syncthreads();   // single-wave WG: NOP barrier + DS fence

        // ---- shared V fragments; O += P(16x32) * V(32x64) for both tiles ----
        v16bf vf[4];
        vf[0] = load_km(Vbase + (size_t)0  * S_, S_, lane, key0);
        vf[1] = load_km(Vbase + (size_t)16 * S_, S_, lane, key0);
        vf[2] = load_km(Vbase + (size_t)32 * S_, S_, lane, key0);
        vf[3] = load_km(Vbase + (size_t)48 * S_, S_, lane, key0);

#pragma unroll
        for (int t = 0; t < 2; ++t) {
            const v16bf pa = load_km(&pLDS[t][0][0], 40, lane, 0);
            o[t][0] = wmma_bf16(pa, vf[0], o[t][0]);
            o[t][1] = wmma_bf16(pa, vf[1], o[t][1]);
            o[t][2] = wmma_bf16(pa, vf[2], o[t][2]);
            o[t][3] = wmma_bf16(pa, vf[3], o[t][3]);
        }
        __syncthreads();
    }

    // ---- epilogue: O / l -> merged-head bf16 [B,S,D] for output GEMM ----
#pragma unroll
    for (int t = 0; t < 2; ++t)
#pragma unroll
        for (int r = 0; r < 8; ++r) {
            const int row = q0[t] + r + 8 * g;
            const float inv = 1.0f / l[t][r];
            const size_t rowoff = ((size_t)b_ * S_ + row) * D_ + h * DEP;
            Attn[rowoff + ci]      = (__bf16)(o[t][0][r] * inv);
            Attn[rowoff + 16 + ci] = (__bf16)(o[t][1][r] * inv);
            Attn[rowoff + 32 + ci] = (__bf16)(o[t][2][r] * inv);
            Attn[rowoff + 48 + ci] = (__bf16)(o[t][3][r] * inv);
        }
}

// ---------------------------------------------------------------------------
// Host launcher
// ---------------------------------------------------------------------------
extern "C" void kernel_launch(void* const* d_in, const int* in_sizes, int n_in,
                              void* d_out, int out_size, void* d_ws, size_t ws_size,
                              hipStream_t stream) {
    (void)in_sizes; (void)n_in; (void)out_size; (void)ws_size;

    const float* q    = (const float*)d_in[0];
    const float* k    = (const float*)d_in[1];
    const float* v    = (const float*)d_in[2];
    const float* mask = (const float*)d_in[3];
    const float* Wq   = (const float*)d_in[4];
    const float* bq   = (const float*)d_in[5];
    const float* Wk   = (const float*)d_in[6];
    const float* bk   = (const float*)d_in[7];
    const float* Wv   = (const float*)d_in[8];
    const float* bv   = (const float*)d_in[9];
    const float* Wo   = (const float*)d_in[10];
    const float* bo   = (const float*)d_in[11];
    float* out        = (float*)d_out;

    // ---- workspace carve-up (all 256B aligned) ----
    size_t off = 0;
    char* ws = (char*)d_ws;
    auto take = [&](size_t bytes) -> void* {
        void* p = ws + off;
        off += (bytes + 255) & ~(size_t)255;
        return p;
    };
    const size_t NE = (size_t)M_ * D_;        // 6,291,456 activations
    const size_t NW = (size_t)D_ * D_;        // 589,824 weight elems

    __bf16* qb   = (__bf16*)take(NE * 2);
    __bf16* kb   = (__bf16*)take(NE * 2);
    __bf16* vb   = (__bf16*)take(NE * 2);
    __bf16* Wqt  = (__bf16*)take(NW * 2);
    __bf16* Wkt  = (__bf16*)take(NW * 2);
    __bf16* Wvt  = (__bf16*)take(NW * 2);
    __bf16* Wot  = (__bf16*)take(NW * 2);
    __bf16* Qh   = (__bf16*)take(NE * 2);     // [bh][s][dep]
    __bf16* Kh   = (__bf16*)take(NE * 2);     // [bh][s][dep]
    __bf16* Vt   = (__bf16*)take(NE * 2);     // [bh][dep][s]
    __bf16* Attn = (__bf16*)take(NE * 2);     // [B,S,D] merged heads

    // ---- 1. precision conversion ----
    {
        const int thr = 256;
        const int blkA = (int)((NE + thr - 1) / thr);
        k_cvt_bf16<<<blkA, thr, 0, stream>>>(q, qb, (int)NE);
        k_cvt_bf16<<<blkA, thr, 0, stream>>>(k, kb, (int)NE);
        k_cvt_bf16<<<blkA, thr, 0, stream>>>(v, vb, (int)NE);
        const int blkW = (int)((NW + thr - 1) / thr);
        k_w_transpose_bf16<<<blkW, thr, 0, stream>>>(Wq, Wqt);
        k_w_transpose_bf16<<<blkW, thr, 0, stream>>>(Wk, Wkt);
        k_w_transpose_bf16<<<blkW, thr, 0, stream>>>(Wv, Wvt);
        k_w_transpose_bf16<<<blkW, thr, 0, stream>>>(Wo, Wot);
    }

    // ---- 2. QKV projections (WMMA GEMMs, 32x32 tiles/wave) ----
    {
        const int tiles  = (M_ / 32) * (D_ / 32);   // 6144
        const int blocks = tiles / 4;               // 1536
        k_gemm<<<blocks, 128, 0, stream>>>(qb, Wqt, bq, Qh, nullptr, 0);
        k_gemm<<<blocks, 128, 0, stream>>>(kb, Wkt, bk, Kh, nullptr, 0);
        k_gemm<<<blocks, 128, 0, stream>>>(vb, Wvt, bv, Vt, nullptr, 1);
    }

    // ---- 3. flash attention (32 query rows per wave) ----
    {
        dim3 grid(S_ / 32, B_ * H_);                // 128 x 24
        k_flash<<<grid, 32, 0, stream>>>(Qh, Kh, Vt, mask, Attn);
    }

    // ---- 4. output projection (f32 + bias) ----
    {
        const int tiles  = (M_ / 32) * (D_ / 32);
        const int blocks = tiles / 4;
        k_gemm<<<blocks, 128, 0, stream>>>(Attn, Wot, bo, nullptr, out, 2);
    }
}
